// GIFNeuron_90314572300868
// MI455X (gfx1250) — matrix operand order repaired and checked by reading, not compiled, not run on hardware
//
#include <hip/hip_runtime.h>

#define BATCHN 32
#define SEQN   1024
#define INDIM  1024
#define HIDN   1024

#define NTILE 128           // hid columns per workgroup (2 tiles per producer wave)
#define MTILE 16            // t rows per chunk (WMMA M)
#define NCHUNK (SEQN / MTILE)
#define WSTR  (INDIM + 8)   // LDS row stride for W tile (bf16 elements)
#define ASTR  (INDIM + 8)   // LDS row stride for A chunk (bf16 elements)
#define SSTR  (NTILE + 4)   // scan buffer row stride (floats)
#define NPROD 128           // 4 producer (GEMM) waves
#define NTHREADS 256        // + 4 consumer (scan) waves

typedef __attribute__((ext_vector_type(16))) __bf16 v16bf;
typedef __attribute__((ext_vector_type(4)))  __bf16 v4bf;
typedef __attribute__((ext_vector_type(8)))  float  v8f;

union FragU { uint4 q[2]; v16bf v; };

// Native RNE converts (v_cvt_[pk_]bf16_f32) + one 8-byte LDS store.
static __device__ __forceinline__ void cvt_store4(__bf16* dst, float4 f) {
    v4bf p = { (__bf16)f.x, (__bf16)f.y, (__bf16)f.z, (__bf16)f.w };
    *(v4bf*)dst = p;
}

// One 16-step GIF scan over a staged 16xNTILE projection tile.
static __device__ __forceinline__ void scan_chunk(const float* __restrict__ inBuf,
                                                  int t0, int c, int h, int b,
                                                  float& v, float& theta, float bias,
                                                  float* __restrict__ spikes) {
    const float DECAYC = 0.9048374180359595f;  // exp(-1/10)
#pragma unroll 4
    for (int r = 0; r < MTILE; ++r) {
        const float cur = inBuf[r * SSTR + c] + bias;
        v = v * DECAYC + cur;
        const float cl = 32.0f * theta;                 // L * theta * 2
        v = fminf(fmaxf(v, -cl), cl);
        float s = floorf(v / (theta + 1e-6f));          // exact IEEE div (matches ref)
        s = fminf(fmaxf(s, 0.0f), 16.0f);               // clip to [0, L]
        v = v - s * theta;
        theta = theta + 0.01f * s - 0.01f * (theta - 1.0f);
        spikes[((size_t)b * SEQN + t0 + r) * HIDN + h] = s;
    }
}

__global__ void __launch_bounds__(NTHREADS)
gif_fused_kernel(const float* __restrict__ X,    // [B, T, IN]
                 const float* __restrict__ W,    // [HID, IN]
                 const float* __restrict__ Bias, // [HID]
                 float* __restrict__ Out)        // spikes | v | theta
{
    extern __shared__ char smem[];
    __bf16* Wlds = (__bf16*)smem;                          // [NTILE][WSTR]
    __bf16* Alds = Wlds + NTILE * WSTR;                    // [MTILE][ASTR]
    float*  scanBuf = (float*)(Alds + MTILE * ASTR);       // [2][MTILE][SSTR]

    const int tid   = threadIdx.x;
    const int lane  = tid & 31;           // wave32
    const int wave  = tid >> 5;           // waves 0-3 produce, 4-7 consume
    const bool producer = tid < NPROD;
    const int hBase = blockIdx.x * NTILE;
    const int b     = blockIdx.y;

    // ---- Stage W tile once (all 8 waves): 128x1024 fp32 -> bf16 LDS
    for (int i = tid * 4; i < NTILE * INDIM; i += NTHREADS * 4) {
        const int row = i >> 10, col = i & (INDIM - 1);
        cvt_store4(Wlds + row * WSTR + col,
                   *(const float4*)(W + (size_t)(hBase + row) * INDIM + col));
    }

    float* spikes = Out;
    float* vOut   = Out + (size_t)BATCHN * SEQN * HIDN;
    float* thOut  = vOut + (size_t)BATCHN * HIDN;

    // ---- Consumer scan state (persistent across all t)
    const int c = tid - NPROD;            // consumer hid column 0..127
    float v = 0.0f, theta = 1.0f, bias = 0.0f;
    if (!producer) bias = Bias[hBase + c];

    // ---- Producer fragment geometry (CDNA5 16x16x32 bf16 VGPR layouts)
    const int arow  = lane & 15;          // A: row M; B: column N; C: N
    const int khalf = lane >> 4;          // selects K-half (this code's contiguous k order)
    const __bf16* aRow  = Alds + arow * ASTR;
    const __bf16* bRow0 = Wlds + (wave * 32 + arow) * WSTR;      // tile 0
    const __bf16* bRow1 = Wlds + (wave * 32 + 16 + arow) * WSTR; // tile 1

    for (int tc = 0; tc < NCHUNK; ++tc) {
        const int t0 = tc * MTILE;

        // ---- ALL waves stage A chunk: 16x1024 fp32 -> bf16 LDS (contiguous rows).
        // Consumers are otherwise idle in this phase; sharing halves its length.
        {
            const float* src = X + ((size_t)b * SEQN + t0) * INDIM;
            for (int i = tid * 4; i < MTILE * INDIM; i += NTHREADS * 4) {
                const float4 x4 = *(const float4*)(src + i);
                if (tc + 1 < NCHUNK)   // warm L2/L0 for the next chunk
                    __builtin_prefetch(src + i + MTILE * INDIM, 0, 1);
                cvt_store4(Alds + (i >> 10) * ASTR + (i & (INDIM - 1)), x4);
            }
        }
        __syncthreads();   // A(tc) ready; buf[tc&1] free (scan of tc-2 done)

        if (producer) {
            // ---- GEMM chunk tc: each wave TWO 16x16 C tiles sharing A fragments.
            // Two independent accumulator chains -> back-to-back WMMAs independent;
            // A loaded once per 2 WMMAs (3 b128 / WMMA).
            v8f acc0 = {0.f, 0.f, 0.f, 0.f, 0.f, 0.f, 0.f, 0.f};
            v8f acc1 = {0.f, 0.f, 0.f, 0.f, 0.f, 0.f, 0.f, 0.f};
            float* outBuf = scanBuf + (tc & 1) * (MTILE * SSTR);
#pragma unroll 4
            for (int kb = 0; kb < INDIM; kb += 32) {
                FragU fa, fb0, fb1;
                // A 16x32: lanes 0-15 K 0..15; lanes 16-31 K 16..31 (per row)
                fa.q[0]  = *(const uint4*)(aRow + kb + khalf * 16);
                fa.q[1]  = *(const uint4*)(aRow + kb + khalf * 16 + 8);
                // B 32x16: lanes 0-15 K 0..15; lanes 16-31 K 16..31 (per column)
                fb0.q[0] = *(const uint4*)(bRow0 + kb + khalf * 16);
                fb0.q[1] = *(const uint4*)(bRow0 + kb + khalf * 16 + 8);
                fb1.q[0] = *(const uint4*)(bRow1 + kb + khalf * 16);
                fb1.q[1] = *(const uint4*)(bRow1 + kb + khalf * 16 + 8);
                acc0 = __builtin_amdgcn_wmma_f32_16x16x32_bf16(
                    false, fa.v, false, fb0.v, (short)0, acc0, false, false);
                acc1 = __builtin_amdgcn_wmma_f32_16x16x32_bf16(
                    false, fa.v, false, fb1.v, (short)0, acc1, false, false);
            }
            // C layout: VGPR r, lanes 0-15 -> M=r, lanes 16-31 -> M=8+r
#pragma unroll
            for (int r = 0; r < 8; ++r) {
                const int m = r + 8 * khalf;
                outBuf[m * SSTR + wave * 32 + arow]      = acc0[r];
                outBuf[m * SSTR + wave * 32 + 16 + arow] = acc1[r];
            }
        } else if (tc > 0) {
            // ---- Consumers scan chunk tc-1 concurrently with GEMM(tc)
            const float* inBuf = scanBuf + ((tc - 1) & 1) * (MTILE * SSTR);
            scan_chunk(inBuf, (tc - 1) * MTILE, c, hBase + c, b, v, theta, bias, spikes);
        }
        __syncthreads();   // GEMM(tc) done; scan(tc-1) done before buf reuse
    }

    // ---- Drain: scan the last chunk, emit final state
    if (!producer) {
        const float* inBuf = scanBuf + ((NCHUNK - 1) & 1) * (MTILE * SSTR);
        scan_chunk(inBuf, (NCHUNK - 1) * MTILE, c, hBase + c, b, v, theta, bias, spikes);
        vOut[(size_t)b * HIDN + hBase + c]  = v;
        thOut[(size_t)b * HIDN + hBase + c] = theta;
    }
}

extern "C" void kernel_launch(void* const* d_in, const int* in_sizes, int n_in,
                              void* d_out, int out_size, void* d_ws, size_t ws_size,
                              hipStream_t stream) {
    const float* x = (const float*)d_in[0];
    const float* W = (const float*)d_in[1];
    const float* b = (const float*)d_in[2];
    float* out = (float*)d_out;

    const size_t smemBytes =
        (size_t)(NTILE * WSTR + MTILE * ASTR) * sizeof(__bf16) +
        (size_t)2 * MTILE * SSTR * sizeof(float);   // ~307 KB < 320 KB/WGP

    (void)hipFuncSetAttribute((const void*)gif_fused_kernel,
                              hipFuncAttributeMaxDynamicSharedMemorySize,
                              (int)smemBytes);

    dim3 grid(HIDN / NTILE, BATCHN);   // 8 x 32 = 256 workgroups
    gif_fused_kernel<<<grid, NTHREADS, smemBytes, stream>>>(x, W, b, out);
}
